// VQNCA_AutoT_61297773248883
// MI455X (gfx1250) — matrix operand order repaired and checked by reading, not compiled, not run on hardware
//
#include <hip/hip_runtime.h>
#include <hip/hip_bf16.h>

// ---------------------------------------------------------------------------
// Types for WMMA fragments (wave32, v_wmma_f32_16x16x32_f16)
// ---------------------------------------------------------------------------
typedef __attribute__((ext_vector_type(16))) _Float16 v16h;
typedef __attribute__((ext_vector_type(8)))  _Float16 f16x8;
typedef __attribute__((ext_vector_type(8)))  float    v8f;

#define NP       262144      // 16 * 128 * 128 pixels
#define HIDDEN   64
#define NCODES   512
#define NSTEPS   5
#define WPB      8           // waves per block (256 threads)
#define VQ_BLOCKS 2048       // NP / (16 px/wave * 8 waves/block)

static __device__ inline v8f wmma_f16(v16h a, v16h b, v8f c) {
  return __builtin_amdgcn_wmma_f32_16x16x32_f16(
      /*neg_a=*/false, a, /*neg_b=*/false, b,
      /*c_mod=*/(short)0, c, /*reuse_a=*/false, /*reuse_b=*/false);
}

// Load a 16x32 f16 A-fragment row slice for this lane.
// base points at (pixel row)*64 + c0 in an NHWC f16 buffer (16B aligned).
// Lane layout: lanes 0-15: K 0..15 ; lanes 16-31: K 16..31
static __device__ inline v16h load_a16(const _Float16* base, int hi) {
  const int b0 = hi * 16;
  const int b1 = hi * 16 + 8;
  f16x8 lo = *(const f16x8*)(base + b0);
  f16x8 hv = *(const f16x8*)(base + b1);
  v16h a;
#pragma unroll
  for (int i = 0; i < 8; ++i) { a[i] = lo[i]; a[i + 8] = hv[i]; }
  return a;
}

static __device__ inline v16h zero_a16() {
  v16h a;
#pragma unroll
  for (int i = 0; i < 16; ++i) a[i] = (_Float16)0.0f;
  return a;
}

// ---------------------------------------------------------------------------
// Weight repack kernels: fragment-ready layout [kt][nt][lane(32)][j(16)]
// frag element: K = kt*32 + (lane>=16)*16 + j , N = nt*16 + (lane&15)
// ---------------------------------------------------------------------------
__global__ __launch_bounds__(256) void repack_conv3x3(
    const float* __restrict__ w, _Float16* __restrict__ frag) {
  int t = blockIdx.x * 256 + threadIdx.x;
  if (t >= 18 * 4 * 32 * 16) return;
  int j = t & 15, lane = (t >> 4) & 31, nt = (t >> 9) & 3, kt = t >> 11; // kt 0..17
  int k = kt * 32 + (lane >> 4) * 16 + j;        // 0..575 : tap*64 + ci
  int n = nt * 16 + (lane & 15);                 // output channel
  int tap = k >> 6, ci = k & 63;
  int ky = tap / 3, kx = tap % 3;
  frag[t] = (_Float16)w[(((size_t)n * 64 + ci) * 3 + ky) * 3 + kx]; // OIHW
}

__global__ __launch_bounds__(256) void repack_1x1(
    const float* __restrict__ w, _Float16* __restrict__ frag) {
  int t = blockIdx.x * 256 + threadIdx.x;
  if (t >= 2 * 4 * 32 * 16) return;
  int j = t & 15, lane = (t >> 4) & 31, nt = (t >> 9) & 3, kt = t >> 11; // kt 0..1
  int k = kt * 32 + (lane >> 4) * 16 + j;
  int n = nt * 16 + (lane & 15);
  frag[t] = (_Float16)w[(size_t)n * 64 + k];     // O x I
}

__global__ __launch_bounds__(256) void repack_codebook(
    const float* __restrict__ cb, _Float16* __restrict__ frag) {
  int t = blockIdx.x * 256 + threadIdx.x;
  if (t >= 2 * 32 * 32 * 16) return;
  int j = t & 15, lane = (t >> 4) & 31, nt = (t >> 9) & 31, kt = (t >> 14) & 1;
  int k = kt * 32 + (lane >> 4) * 16 + j;        // channel
  int n = nt * 16 + (lane & 15);                 // code index
  frag[t] = (_Float16)cb[(size_t)n * 64 + k];    // B[k][n] = codebook[n][k]
}

__global__ __launch_bounds__(256) void codebook_norms(
    const float* __restrict__ cb, float* __restrict__ nrm) {
  int n = blockIdx.x * 256 + threadIdx.x;
  if (n >= NCODES) return;
  float s = 0.f;
#pragma unroll
  for (int c = 0; c < 64; ++c) { float v = cb[(size_t)n * 64 + c]; s += v * v; }
  nrm[n] = s;
}

__global__ void zero_loss(float* loss) { loss[0] = 0.f; }

// ---------------------------------------------------------------------------
// Stem: 3x3 conv (1 -> 64) + ReLU, writes NHWC f32 + f16 mirror
// ---------------------------------------------------------------------------
__global__ __launch_bounds__(256) void stem_relu(
    const float* __restrict__ x, const float* __restrict__ w,
    const float* __restrict__ b, float* __restrict__ sF32,
    _Float16* __restrict__ sF16) {
  int p = blockIdx.x * 256 + threadIdx.x;
  if (p >= NP) return;
  int xx = p & 127, yy = (p >> 7) & 127, bb = p >> 14;
  float xv[9];
#pragma unroll
  for (int t = 0; t < 9; ++t) {
    int dy = t / 3 - 1, dx = t % 3 - 1;
    int X = xx + dx, Y = yy + dy;
    xv[t] = ((unsigned)X < 128u && (unsigned)Y < 128u)
                ? x[(size_t)bb * 16384 + Y * 128 + X] : 0.f;
  }
#pragma unroll 4
  for (int c = 0; c < 64; ++c) {
    float acc = b[c];
#pragma unroll
    for (int t = 0; t < 9; ++t) acc = fmaf(w[c * 9 + t], xv[t], acc);
    acc = fmaxf(acc, 0.f);
    size_t idx = (size_t)p * 64 + c;
    sF32[idx] = acc;
    sF16[idx] = (_Float16)acc;
  }
}

// ---------------------------------------------------------------------------
// up1: implicit-GEMM 3x3 conv 64->64 + ReLU via WMMA.
// Each wave: 16 pixels x 64 outputs, K = 576 (18 WMMA steps x 4 N-tiles).
// ---------------------------------------------------------------------------
__global__ __launch_bounds__(256) void conv3x3_wmma(
    const _Float16* __restrict__ inF16, const _Float16* __restrict__ frag,
    const float* __restrict__ bias, _Float16* __restrict__ outF16) {
  const int lane = threadIdx.x & 31;
  const int wave = blockIdx.x * WPB + (threadIdx.x >> 5);
  const int pix0 = wave * 16;
  const int m = lane & 15, hi = lane >> 4;
  const int p = pix0 + m;
  const int xx = p & 127, yy = (p >> 7) & 127;

  v8f acc[4];
#pragma unroll
  for (int nt = 0; nt < 4; ++nt)
#pragma unroll
    for (int r = 0; r < 8; ++r) acc[nt][r] = 0.f;

  for (int kk = 0; kk < 18; ++kk) {
    const int tap = kk >> 1, c0 = (kk & 1) * 32;
    const int dy = tap / 3 - 1, dx = tap % 3 - 1;
    const bool valid = ((unsigned)(xx + dx) < 128u) && ((unsigned)(yy + dy) < 128u);
    v16h a;
    if (valid)
      a = load_a16(inF16 + (size_t)(p + dy * 128 + dx) * 64 + c0, hi);
    else
      a = zero_a16();
    const _Float16* fp = frag + (size_t)kk * 4 * 512;
#pragma unroll
    for (int nt = 0; nt < 4; ++nt) {
      v16h b = *(const v16h*)(fp + ((size_t)nt * 32 + lane) * 16);
      acc[nt] = wmma_f16(a, b, acc[nt]);
    }
  }

#pragma unroll
  for (int nt = 0; nt < 4; ++nt) {
    const int n = nt * 16 + m;
    const float bn = bias[n];
#pragma unroll
    for (int r = 0; r < 8; ++r) {
      const int pr = pix0 + hi * 8 + r;
      float v = fmaxf(acc[nt][r] + bn, 0.f);
      outF16[(size_t)pr * 64 + n] = (_Float16)v;
    }
  }
}

// ---------------------------------------------------------------------------
// Fused: delta = hidden @ W2^T + b2 ; tau = state @ Wt^T + bt ;
//        state = sigmoid(tau)*state + (1-sigmoid(tau))*delta  (f32 + f16)
// ---------------------------------------------------------------------------
__global__ __launch_bounds__(256) void step_mid_wmma(
    const _Float16* __restrict__ hidF16, const _Float16* __restrict__ stF16,
    const _Float16* __restrict__ frag2, const _Float16* __restrict__ fragT,
    const float* __restrict__ b2, const float* __restrict__ bt,
    float* __restrict__ sF32, _Float16* __restrict__ sF16out) {
  const int lane = threadIdx.x & 31;
  const int wave = blockIdx.x * WPB + (threadIdx.x >> 5);
  const int pix0 = wave * 16;
  const int m = lane & 15, hi = lane >> 4;
  const int p = pix0 + m;

  v8f dacc[4], tacc[4];
#pragma unroll
  for (int nt = 0; nt < 4; ++nt)
#pragma unroll
    for (int r = 0; r < 8; ++r) { dacc[nt][r] = 0.f; tacc[nt][r] = 0.f; }

#pragma unroll
  for (int kt = 0; kt < 2; ++kt) {
    const int c0 = kt * 32;
    v16h ah = load_a16(hidF16 + (size_t)p * 64 + c0, hi);
    v16h as = load_a16(stF16 + (size_t)p * 64 + c0, hi);
#pragma unroll
    for (int nt = 0; nt < 4; ++nt) {
      v16h bd = *(const v16h*)(frag2 + (((size_t)kt * 4 + nt) * 32 + lane) * 16);
      dacc[nt] = wmma_f16(ah, bd, dacc[nt]);
      v16h bb = *(const v16h*)(fragT + (((size_t)kt * 4 + nt) * 32 + lane) * 16);
      tacc[nt] = wmma_f16(as, bb, tacc[nt]);
    }
  }

#pragma unroll
  for (int nt = 0; nt < 4; ++nt) {
    const int n = nt * 16 + m;
    const float db = b2[n], tb = bt[n];
#pragma unroll
    for (int r = 0; r < 8; ++r) {
      const int pr = pix0 + hi * 8 + r;
      const size_t idx = (size_t)pr * 64 + n;
      float delta = dacc[nt][r] + db;
      float tl    = tacc[nt][r] + tb;
      float beta  = 1.f / (1.f + __expf(-tl));
      float ns    = beta * sF32[idx] + (1.f - beta) * delta;
      sF32[idx]    = ns;
      sF16out[idx] = (_Float16)ns;
    }
  }
}

// ---------------------------------------------------------------------------
// VQ: nearest-code via WMMA (score = |c|^2 - 2 z.c ; |z|^2 is row-constant),
// cross-lane argmin, codebook gather, straight-through write, loss partials.
// ---------------------------------------------------------------------------
__global__ __launch_bounds__(256) void vq_wmma(
    const _Float16* __restrict__ cbFrag, const float* __restrict__ cbNrm,
    const float* __restrict__ codebook, float* __restrict__ sF32,
    _Float16* __restrict__ sF16, float* __restrict__ partials) {
  __shared__ float red[256];
  const int lane = threadIdx.x & 31;
  const int wave = blockIdx.x * WPB + (threadIdx.x >> 5);
  const int pix0 = wave * 16;
  const int m = lane & 15, hi = lane >> 4;
  const int p = pix0 + m;

  v16h a0 = load_a16(sF16 + (size_t)p * 64 + 0, hi);
  v16h a1 = load_a16(sF16 + (size_t)p * 64 + 32, hi);

  float minv[8];
  int   mini[8];
#pragma unroll
  for (int r = 0; r < 8; ++r) { minv[r] = 3.4e38f; mini[r] = 0; }

  for (int nt = 0; nt < 32; ++nt) {
    v8f dot;
#pragma unroll
    for (int r = 0; r < 8; ++r) dot[r] = 0.f;
    v16h b0 = *(const v16h*)(cbFrag + (((size_t)0 * 32 + nt) * 32 + lane) * 16);
    dot = wmma_f16(a0, b0, dot);
    v16h b1 = *(const v16h*)(cbFrag + (((size_t)1 * 32 + nt) * 32 + lane) * 16);
    dot = wmma_f16(a1, b1, dot);
    const int n = nt * 16 + m;
    const float cn = cbNrm[n];
#pragma unroll
    for (int r = 0; r < 8; ++r) {
      float score = cn - 2.f * dot[r];
      if (score < minv[r]) { minv[r] = score; mini[r] = n; }
    }
  }

  // argmin across each 16-lane column group (rows stay in-lane-group)
#pragma unroll
  for (int r = 0; r < 8; ++r) {
#pragma unroll
    for (int off = 1; off <= 8; off <<= 1) {
      float ov = __shfl_xor(minv[r], off, 16);
      int   oi = __shfl_xor(mini[r], off, 16);
      if (ov < minv[r] || (ov == minv[r] && oi < mini[r])) {
        minv[r] = ov; mini[r] = oi;
      }
    }
  }

  // Gather codebook rows, straight-through assign, commit/codebook loss.
  float lsum = 0.f;
#pragma unroll
  for (int i = 0; i < 16; ++i) {
    const int r = i & 7, srcHalf = i >> 3;
    const int code = __shfl(mini[r], srcHalf * 16, 32);
    const int pi = pix0 + i;
    const float* cb = codebook + (size_t)code * 64;
#pragma unroll
    for (int u = 0; u < 2; ++u) {
      const int c = lane * 2 + u;
      const size_t idx = (size_t)pi * 64 + c;
      float zq = cb[c];
      float d  = zq - sF32[idx];
      lsum += d * d;
      sF32[idx] = zq;
      sF16[idx] = (_Float16)zq;
    }
  }

  // deterministic block tree reduction
  red[threadIdx.x] = lsum;
  __syncthreads();
#pragma unroll
  for (int s = 128; s > 0; s >>= 1) {
    if (threadIdx.x < s) red[threadIdx.x] += red[threadIdx.x + s];
    __syncthreads();
  }
  if (threadIdx.x == 0) partials[blockIdx.x] = red[0];
}

__global__ __launch_bounds__(256) void reduce_partials(
    const float* __restrict__ partials, float* __restrict__ loss) {
  __shared__ float red[256];
  float s = 0.f;
  for (int i = threadIdx.x; i < VQ_BLOCKS; i += 256) s += partials[i];
  red[threadIdx.x] = s;
  __syncthreads();
#pragma unroll
  for (int k = 128; k > 0; k >>= 1) {
    if (threadIdx.x < k) red[threadIdx.x] += red[threadIdx.x + k];
    __syncthreads();
  }
  if (threadIdx.x == 0) loss[0] += red[0];
}

// ---------------------------------------------------------------------------
// Decoder: 1x1 conv 64 -> 1 + sigmoid
// ---------------------------------------------------------------------------
__global__ __launch_bounds__(256) void dec_sigmoid(
    const float* __restrict__ sF32, const float* __restrict__ w,
    const float* __restrict__ b, float* __restrict__ out) {
  int p = blockIdx.x * 256 + threadIdx.x;
  if (p >= NP) return;
  float acc = b[0];
#pragma unroll
  for (int c = 0; c < 64; ++c) acc = fmaf(w[c], sF32[(size_t)p * 64 + c], acc);
  out[p] = 1.f / (1.f + __expf(-acc));
}

__global__ void finalize_loss(const float* __restrict__ loss,
                              const int* __restrict__ n_steps,
                              float* __restrict__ out_scalar) {
  float ns = (float)(*n_steps);
  // vq_loss per step = codebook + 0.25*commit = 1.25 * mean((zq - z)^2)
  out_scalar[0] = 1.25f * loss[0] / (16777216.0f * ns);
}

// ---------------------------------------------------------------------------
// Host-side orchestration
// ---------------------------------------------------------------------------
extern "C" void kernel_launch(void* const* d_in, const int* in_sizes, int n_in,
                              void* d_out, int out_size, void* d_ws, size_t ws_size,
                              hipStream_t stream) {
  (void)in_sizes; (void)n_in; (void)out_size; (void)ws_size;
  const float* x       = (const float*)d_in[0];
  const float* stem_w  = (const float*)d_in[1];
  const float* stem_b  = (const float*)d_in[2];
  const float* up1_w   = (const float*)d_in[3];
  const float* up1_b   = (const float*)d_in[4];
  const float* up2_w   = (const float*)d_in[5];
  const float* up2_b   = (const float*)d_in[6];
  const float* tau_w   = (const float*)d_in[7];
  const float* tau_b   = (const float*)d_in[8];
  const float* codebook= (const float*)d_in[9];
  const float* dec_w   = (const float*)d_in[10];
  const float* dec_b   = (const float*)d_in[11];
  const int*   n_steps = (const int*)d_in[12];
  float* out = (float*)d_out;

  // Workspace carve-out (256B aligned slices)
  char* ws = (char*)d_ws;
  size_t off = 0;
  auto take = [&](size_t bytes) -> void* {
    void* p = ws + off;
    off = (off + bytes + 255) & ~(size_t)255;
    return p;
  };
  float*     stateF32 = (float*)take((size_t)NP * 64 * 4);     // 64 MiB
  _Float16*  stateF16 = (_Float16*)take((size_t)NP * 64 * 2);  // 32 MiB
  _Float16*  hidF16   = (_Float16*)take((size_t)NP * 64 * 2);  // 32 MiB
  _Float16*  fragUp1  = (_Float16*)take((size_t)18 * 4 * 512 * 2);
  _Float16*  fragUp2  = (_Float16*)take((size_t)2 * 4 * 512 * 2);
  _Float16*  fragTau  = (_Float16*)take((size_t)2 * 4 * 512 * 2);
  _Float16*  fragCB   = (_Float16*)take((size_t)2 * 32 * 512 * 2);
  float*     cbNrm    = (float*)take((size_t)NCODES * 4);
  float*     partials = (float*)take((size_t)VQ_BLOCKS * 4);
  float*     lossAcc  = (float*)take(4);

  // one-time per-launch prep (deterministic, graph-capturable)
  zero_loss<<<1, 1, 0, stream>>>(lossAcc);
  repack_conv3x3<<<144, 256, 0, stream>>>(up1_w, fragUp1);
  repack_1x1<<<16, 256, 0, stream>>>(up2_w, fragUp2);
  repack_1x1<<<16, 256, 0, stream>>>(tau_w, fragTau);
  repack_codebook<<<128, 256, 0, stream>>>(codebook, fragCB);
  codebook_norms<<<2, 256, 0, stream>>>(codebook, cbNrm);

  stem_relu<<<NP / 256, 256, 0, stream>>>(x, stem_w, stem_b, stateF32, stateF16);

  for (int s = 0; s < NSTEPS; ++s) {
    conv3x3_wmma<<<VQ_BLOCKS, 256, 0, stream>>>(stateF16, fragUp1, up1_b, hidF16);
    step_mid_wmma<<<VQ_BLOCKS, 256, 0, stream>>>(hidF16, stateF16, fragUp2, fragTau,
                                                 up2_b, tau_b, stateF32, stateF16);
    vq_wmma<<<VQ_BLOCKS, 256, 0, stream>>>(fragCB, cbNrm, codebook,
                                           stateF32, stateF16, partials);
    reduce_partials<<<1, 256, 0, stream>>>(partials, lossAcc);
  }

  dec_sigmoid<<<NP / 256, 256, 0, stream>>>(stateF32, dec_w, dec_b, out);
  finalize_loss<<<1, 1, 0, stream>>>(lossAcc, n_steps, out + NP);
}
